// MultiScaleRetention_30245159698613
// MI455X (gfx1250) — matrix operand order repaired and checked by reading, not compiled, not run on hardware
//
#include <hip/hip_runtime.h>
#include <hip/hip_bf16.h>
#include <stdint.h>

// ---------------- problem constants ----------------
#define B_   4
#define T_   2048
#define DM   1024      // d_model
#define QKD  1024      // qk dim
#define VD   2048      // v dim
#define NH   8
#define DK   128       // per-head qk
#define DV   256       // per-head v
#define QKVG_LD 6144   // q(1024) | k(1024) | v(2048) | gate(2048)

typedef __attribute__((ext_vector_type(16))) __bf16 bf16x16;
typedef __attribute__((ext_vector_type(8)))  float  floatx8;

__device__ __forceinline__ unsigned short f2bf(float f) {
  unsigned int x = __float_as_uint(f);
  x += 0x7FFFu + ((x >> 16) & 1u);          // round-to-nearest-even
  return (unsigned short)(x >> 16);
}
__device__ __forceinline__ float bf2f(unsigned short b) {
  return __uint_as_float(((unsigned int)b) << 16);
}
__device__ __forceinline__ floatx8 zero8() {
  floatx8 z;
#pragma unroll
  for (int i = 0; i < 8; ++i) z[i] = 0.0f;
  return z;
}

union FragU { uint4 u[2]; bf16x16 v; };

// A fragment (16x32 bf16, row-major tile [m][k], stride in elements).
// ISA layout: lane<16 -> K {k0..k0+7, k0+16..k0+23}; lane>=16 -> +8.
__device__ __forceinline__ bf16x16 frag_a(const unsigned short* __restrict__ base,
                                          int row, int stride, int kk, int lane) {
  const int k0 = kk + ((lane >> 4) << 3);
  const unsigned short* p = base + row * stride + k0;
  FragU f;
  f.u[0] = *reinterpret_cast<const uint4*>(p);
  f.u[1] = *reinterpret_cast<const uint4*>(p + 16);
  return f.v;
}
// B fragment (32x16 bf16) read from a column-major tile Bt[n][k].
// ISA layout: lane<16 -> K {k0..k0+7, k0+16..k0+23}; lane>=16 -> +8.
__device__ __forceinline__ bf16x16 frag_b(const unsigned short* __restrict__ base,
                                          int col, int stride, int kk, int lane) {
  const int kb = kk + ((lane >> 4) << 3);
  const unsigned short* p = base + col * stride + kb;
  FragU f;
  f.u[0] = *reinterpret_cast<const uint4*>(p);
  f.u[1] = *reinterpret_cast<const uint4*>(p + 16);
  return f.v;
}
__device__ __forceinline__ floatx8 wmma_bf16(bf16x16 a, bf16x16 b, floatx8 c) {
  return __builtin_amdgcn_wmma_f32_16x16x32_bf16(false, a, false, b, (short)0, c,
                                                 false, false);
}

// ---------------- 0) f32 -> bf16 convert ----------------
__global__ void cvt_bf16_kernel(const float* __restrict__ src,
                                unsigned short* __restrict__ dst, int n) {
  int i = blockIdx.x * blockDim.x + threadIdx.x;
  if (i < n) dst[i] = f2bf(src[i]);
}

// ---------------- 1) generic bf16 GEMM: C = A(bf16) @ B(f32) ----------------
// block tile 128x128, BK=32, 256 threads = 8 waves, each wave 2x4 WMMA tiles.
template <bool OUT_BF16>
__global__ __launch_bounds__(256) void gemm_bf16_kernel(
    const unsigned short* __restrict__ A, int lda,
    const float* __restrict__ B, int ldb,
    void* __restrict__ Cout, int ldc,
    int M, int N, int K) {
  __shared__ __align__(16) unsigned short As[128 * 32];   // [m][k]
  __shared__ __align__(16) unsigned short Bt[128 * 32];   // [n][k] (transposed)

  const int tid  = threadIdx.x;
  const int lane = tid & 31;
  const int wave = tid >> 5;
  const int m0 = blockIdx.y * 128;
  const int n0 = blockIdx.x * 128;
  const int wm = (wave & 3) * 32;   // 2 m-tiles
  const int wn = (wave >> 2) * 64;  // 4 n-tiles

  floatx8 acc[2][4];
#pragma unroll
  for (int i = 0; i < 2; ++i)
#pragma unroll
    for (int j = 0; j < 4; ++j) acc[i][j] = zero8();

  for (int kk = 0; kk < K; kk += 32) {
    { // A tile load: 2 threads/row, 16 bf16 each
      const int r = tid >> 1;
      const int p = (tid & 1) * 16;
      const uint4* src =
          reinterpret_cast<const uint4*>(A + (size_t)(m0 + r) * lda + kk + p);
      uint4* dst = reinterpret_cast<uint4*>(&As[r * 32 + p]);
      dst[0] = src[0];
      dst[1] = src[1];
      if (kk + 32 < K)
        __builtin_prefetch(A + (size_t)(m0 + r) * lda + kk + 32 + p, 0, 1);
    }
    { // B tile load (f32 -> bf16, store transposed)
      const int kr = tid >> 3;
      const int p  = (tid & 7) * 16;
      const float4* src =
          reinterpret_cast<const float4*>(B + (size_t)(kk + kr) * ldb + n0 + p);
      float4 vv[4];
#pragma unroll
      for (int q = 0; q < 4; ++q) vv[q] = src[q];
      const float* cf = reinterpret_cast<const float*>(vv);
#pragma unroll
      for (int j = 0; j < 16; ++j) Bt[(p + j) * 32 + kr] = f2bf(cf[j]);
      if (kk + 32 < K)
        __builtin_prefetch(B + (size_t)(kk + 32 + kr) * ldb + n0 + p, 0, 1);
    }
    __syncthreads();

    bf16x16 af[2], bf[4];
#pragma unroll
    for (int i = 0; i < 2; ++i)
      af[i] = frag_a(As, wm + i * 16 + (lane & 15), 32, 0, lane);
#pragma unroll
    for (int j = 0; j < 4; ++j)
      bf[j] = frag_b(Bt, wn + j * 16 + (lane & 15), 32, 0, lane);
#pragma unroll
    for (int i = 0; i < 2; ++i)
#pragma unroll
      for (int j = 0; j < 4; ++j) acc[i][j] = wmma_bf16(af[i], bf[j], acc[i][j]);
    __syncthreads();
  }

#pragma unroll
  for (int i = 0; i < 2; ++i)
#pragma unroll
    for (int j = 0; j < 4; ++j)
#pragma unroll
      for (int r = 0; r < 8; ++r) {
        const int row = m0 + wm + i * 16 + r + ((lane >> 4) << 3);
        const int col = n0 + wn + j * 16 + (lane & 15);
        const float v = acc[i][j][r];
        if (OUT_BF16)
          reinterpret_cast<unsigned short*>(Cout)[(size_t)row * ldc + col] = f2bf(v);
        else
          reinterpret_cast<float*>(Cout)[(size_t)row * ldc + col] = v;
      }
}

// ---------------- 2) xPos rotary on q,k (in place, bf16) ----------------
__global__ void rotary_kernel(unsigned short* __restrict__ qkvg) {
  const int idx = blockIdx.x * blockDim.x + threadIdx.x;  // B*T*512 pairs
  if (idx >= B_ * T_ * (QKD / 2)) return;
  const int i   = idx & 511;        // pair index within head-dim
  const int bt  = idx >> 9;
  const int pos = bt & (T_ - 1);
  const float fi = (float)i;
  const float base_scale = (2.0f * fi + 0.4f * 1024.0f) * (1.0f / (1.4f * 1024.0f));
  const float scale = exp2f(log2f(base_scale) * ((float)pos) * (1.0f / 512.0f));
  const float inv_freq = exp2f(-(fi * (1.0f / 512.0f)) * 13.287712379549449f);
  float sn, cs;
  sincosf((float)pos * inv_freq, &sn, &cs);

  unsigned short* qp = qkvg + (size_t)bt * QKVG_LD + 2 * i;
  unsigned short* kp = qp + QKD;
  {
    const float c = cs * scale, s = sn * scale;
    const float x1 = bf2f(qp[0]), x2 = bf2f(qp[1]);
    qp[0] = f2bf(x1 * c - x2 * s);
    qp[1] = f2bf(x2 * c + x1 * s);
  }
  {
    const float inv = 1.0f / scale;
    const float c = cs * inv, s = sn * inv;
    const float x1 = bf2f(kp[0]), x2 = bf2f(kp[1]);
    kp[0] = f2bf(x1 * c - x2 * s);
    kp[1] = f2bf(x2 * c + x1 * s);
  }
}

// ---------------- 3) SiLU on gate (in place, bf16) ----------------
__global__ void silu_kernel(unsigned short* __restrict__ qkvg) {
  const int idx = blockIdx.x * blockDim.x + threadIdx.x;  // B*T*VD
  if (idx >= B_ * T_ * VD) return;
  const int bt = idx >> 11;
  const int c  = idx & (VD - 1);
  unsigned short* p = qkvg + (size_t)bt * QKVG_LD + 2 * QKD + VD + c;
  const float x = bf2f(*p);
  *p = f2bf(x / (1.0f + expf(-x)));
}

// ---------------- 4) retention: out = (decay ⊙ QK^T) V ----------------
// grid (T/64, NH*2, B): 64 q-rows x 128 dv-cols per block, causal key loop.
__global__ __launch_bounds__(256) void retention_kernel(
    const unsigned short* __restrict__ qkvg, const float* __restrict__ decay,
    float* __restrict__ ret) {
  __shared__ __align__(16) unsigned short Qs[64 * 128];  // [t][d]
  __shared__ __align__(16) unsigned short Ks[64 * 128];  // [s][d]
  __shared__ __align__(16) unsigned short Vt[128 * 64];  // [dv][s]
  __shared__ __align__(16) unsigned short Ss[64 * 64];   // [t][s] bf16 scores

  const int b = blockIdx.z;
  const int h = blockIdx.y >> 1;
  const int half = blockIdx.y & 1;
  const int qb = blockIdx.x;
  const int qbase = qb * 64;
  const int tid = threadIdx.x, lane = tid & 31, wave = tid >> 5;
  const float log2d = log2f(decay[h]);
  const float inv_sqrt_dk = 0.08838834764831845f;  // 1/sqrt(128)

  const unsigned short* qcol = qkvg + h * DK;
  const unsigned short* kcol = qkvg + QKD + h * DK;
  const unsigned short* vcol = qkvg + 2 * QKD + h * DV + half * 128;

  { // load Q tile (once)
    const int r = tid >> 2, p = (tid & 3) * 32;
    const uint4* src = reinterpret_cast<const uint4*>(
        qcol + (size_t)(b * T_ + qbase + r) * QKVG_LD + p);
    uint4* dst = reinterpret_cast<uint4*>(&Qs[r * 128 + p]);
#pragma unroll
    for (int q = 0; q < 4; ++q) dst[q] = src[q];
  }

  // GEMM2 accumulators: wave -> m-tile (wave>>1), 4 n-tiles at (wave&1)*64
  const int m2 = wave >> 1;
  const int nb2 = (wave & 1) * 64;
  floatx8 acc[4];
#pragma unroll
  for (int j = 0; j < 4; ++j) acc[j] = zero8();

  for (int sb = 0; sb <= qb; ++sb) {
    const int sbase = sb * 64;
    { // K tile
      const int r = tid >> 2, p = (tid & 3) * 32;
      const uint4* src = reinterpret_cast<const uint4*>(
          kcol + (size_t)(b * T_ + sbase + r) * QKVG_LD + p);
      uint4* dst = reinterpret_cast<uint4*>(&Ks[r * 128 + p]);
#pragma unroll
      for (int q = 0; q < 4; ++q) dst[q] = src[q];
    }
    { // V tile, transposed into [dv][s]
      const int r = tid >> 2, p = (tid & 3) * 32;
      const uint4* src = reinterpret_cast<const uint4*>(
          vcol + (size_t)(b * T_ + sbase + r) * QKVG_LD + p);
      uint4 tmp[4];
#pragma unroll
      for (int q = 0; q < 4; ++q) tmp[q] = src[q];
      const unsigned short* e = reinterpret_cast<const unsigned short*>(tmp);
#pragma unroll
      for (int c = 0; c < 32; ++c) Vt[(p + c) * 64 + r] = e[c];
    }
    __syncthreads();

    // GEMM1: S = Q K^T (each wave: 2 of the 16 16x16 tiles)
#pragma unroll
    for (int tI = 0; tI < 2; ++tI) {
      const int tile = wave * 2 + tI;
      const int mt = tile >> 2, nt = tile & 3;
      floatx8 s = zero8();
#pragma unroll
      for (int kk = 0; kk < 128; kk += 32) {
        bf16x16 a  = frag_a(Qs, mt * 16 + (lane & 15), 128, kk, lane);
        bf16x16 bb = frag_b(Ks, nt * 16 + (lane & 15), 128, kk, lane);
        s = wmma_bf16(a, bb, s);
      }
      const int colS = nt * 16 + (lane & 15);
      const int sg = sbase + colS;
#pragma unroll
      for (int r = 0; r < 8; ++r) {
        const int rowS = mt * 16 + r + ((lane >> 4) << 3);
        const int tg = qbase + rowS;
        const int d = tg - sg;
        float f = 0.0f;
        if (d >= 0) f = s[r] * exp2f(log2d * (float)d) * inv_sqrt_dk;
        Ss[rowS * 64 + colS] = f2bf(f);
      }
    }
    __syncthreads();

    // GEMM2: out += S V
#pragma unroll
    for (int j = 0; j < 4; ++j) {
      const int ncol = nb2 + j * 16;
#pragma unroll
      for (int kk = 0; kk < 64; kk += 32) {
        bf16x16 a  = frag_a(Ss, m2 * 16 + (lane & 15), 64, kk, lane);
        bf16x16 bb = frag_b(Vt, ncol + (lane & 15), 64, kk, lane);
        acc[j] = wmma_bf16(a, bb, acc[j]);
      }
    }
    __syncthreads();
  }

#pragma unroll
  for (int j = 0; j < 4; ++j)
#pragma unroll
    for (int r = 0; r < 8; ++r) {
      const int t  = qbase + m2 * 16 + r + ((lane >> 4) << 3);
      const int dv = half * 128 + nb2 + j * 16 + (lane & 15);
      ret[((size_t)(b * T_ + t)) * VD + h * DV + dv] = acc[j][r];
    }
}

// ---------------- 5) curr_kv = sum_t decay^(T-1-t) k_t ⊗ v_t ----------------
// grid (2, NH, B): 128(qk) x 128(half of dv) per block.
__global__ __launch_bounds__(256) void currkv_kernel(
    const unsigned short* __restrict__ qkvg, const float* __restrict__ decay,
    float* __restrict__ out) {
  __shared__ __align__(16) unsigned short At[128 * 32];  // [d][t], decay-weighted k^T
  __shared__ __align__(16) unsigned short Vt[128 * 32];  // [dv][t]

  const int b = blockIdx.z, h = blockIdx.y, half = blockIdx.x;
  const int nbase = half * 128;
  const int tid = threadIdx.x, lane = tid & 31, wave = tid >> 5;
  const float log2d = log2f(decay[h]);

  floatx8 acc[8];
#pragma unroll
  for (int j = 0; j < 8; ++j) acc[j] = zero8();

  for (int t0 = 0; t0 < T_; t0 += 32) {
    { // k block: load, weight by decay^(T-1-t), store transposed
      const int tr = tid >> 3, p = (tid & 7) * 16;
      const uint4* src = reinterpret_cast<const uint4*>(
          qkvg + QKD + h * DK + (size_t)(b * T_ + t0 + tr) * QKVG_LD + p);
      uint4 tmp[2];
      tmp[0] = src[0];
      tmp[1] = src[1];
      const unsigned short* e = reinterpret_cast<const unsigned short*>(tmp);
      const float w = exp2f(log2d * (float)(T_ - 1 - (t0 + tr)));
#pragma unroll
      for (int j = 0; j < 16; ++j) At[(p + j) * 32 + tr] = f2bf(bf2f(e[j]) * w);
    }
    { // v block transposed
      const int tr = tid >> 3, p = (tid & 7) * 16;
      const uint4* src = reinterpret_cast<const uint4*>(
          qkvg + 2 * QKD + h * DV + nbase + (size_t)(b * T_ + t0 + tr) * QKVG_LD + p);
      uint4 tmp[2];
      tmp[0] = src[0];
      tmp[1] = src[1];
      const unsigned short* e = reinterpret_cast<const unsigned short*>(tmp);
#pragma unroll
      for (int j = 0; j < 16; ++j) Vt[(p + j) * 32 + tr] = e[j];
    }
    __syncthreads();

    bf16x16 a = frag_a(At, wave * 16 + (lane & 15), 32, 0, lane);
#pragma unroll
    for (int j = 0; j < 8; ++j) {
      bf16x16 bb = frag_b(Vt, j * 16 + (lane & 15), 32, 0, lane);
      acc[j] = wmma_bf16(a, bb, acc[j]);
    }
    __syncthreads();
  }

#pragma unroll
  for (int j = 0; j < 8; ++j)
#pragma unroll
    for (int r = 0; r < 8; ++r) {
      const int qk  = wave * 16 + r + ((lane >> 4) << 3);
      const int col = nbase + j * 16 + (lane & 15);
      out[((size_t)((b * NH + h) * DK + qk)) * DV + col] = acc[j][r];
    }
}

// ---------------- 6) GroupNorm + gate multiply ----------------
// grid = B*T blocks; wave w normalizes head w's 256 values.
__global__ __launch_bounds__(256) void gn_gate_kernel(
    const float* __restrict__ ret, const unsigned short* __restrict__ qkvg,
    unsigned short* __restrict__ ng) {
  const int bt = blockIdx.x;
  const int wave = threadIdx.x >> 5, lane = threadIdx.x & 31;
  const float* g = ret + (size_t)bt * VD + wave * DV;
  float x[8], s = 0.0f, s2 = 0.0f;
#pragma unroll
  for (int i = 0; i < 8; ++i) {
    x[i] = g[lane * 8 + i];
    s += x[i];
    s2 += x[i] * x[i];
  }
#pragma unroll
  for (int off = 16; off > 0; off >>= 1) {
    s  += __shfl_xor(s, off, 32);
    s2 += __shfl_xor(s2, off, 32);
  }
  const float mu = s * (1.0f / 256.0f);
  const float var = s2 * (1.0f / 256.0f) - mu * mu;
  const float rstd = rsqrtf(var + 1e-5f);
  const unsigned short* gate = qkvg + (size_t)bt * QKVG_LD + 2 * QKD + VD + wave * DV;
  unsigned short* o = ng + (size_t)bt * VD + wave * DV;
#pragma unroll
  for (int i = 0; i < 8; ++i) {
    const int idx = lane * 8 + i;
    o[idx] = f2bf((x[i] - mu) * rstd * bf2f(gate[idx]));
  }
}

// ---------------- launcher ----------------
extern "C" void kernel_launch(void* const* d_in, const int* in_sizes, int n_in,
                              void* d_out, int out_size, void* d_ws, size_t ws_size,
                              hipStream_t stream) {
  (void)in_sizes; (void)n_in; (void)out_size; (void)ws_size;
  const float* hidden = (const float*)d_in[0];   // (4,2048,1024)
  const float* W_qkv  = (const float*)d_in[1];   // (1024,4096)
  const float* W_g    = (const float*)d_in[2];   // (1024,2048)
  const float* W_proj = (const float*)d_in[3];   // (2048,1024)
  const float* decay  = (const float*)d_in[4];   // (8,)
  float* out = (float*)d_out;                    // proj_out(8M) | curr_kv(1M)

  char* ws = (char*)d_ws;
  unsigned short* hid  = (unsigned short*)(ws);                          // 16 MB
  unsigned short* qkvg = (unsigned short*)(ws + (size_t)(16u << 20));    // 96 MB
  float* ret           = (float*)(ws + (size_t)(112u << 20));            // 64 MB
  unsigned short* ng   = (unsigned short*)(ws + (size_t)(176u << 20));   // 32 MB

  const int M = B_ * T_;  // 8192

  // 0) X -> bf16
  cvt_bf16_kernel<<<(M * DM) / 256, 256, 0, stream>>>(hidden, hid, M * DM);

  // 1) qkv = X @ W_qkv ; gate_raw = X @ W_g  (bf16 out, interleaved ldc=6144)
  gemm_bf16_kernel<true><<<dim3(4096 / 128, M / 128), 256, 0, stream>>>(
      hid, DM, W_qkv, 4096, qkvg, QKVG_LD, M, 4096, DM);
  gemm_bf16_kernel<true><<<dim3(2048 / 128, M / 128), 256, 0, stream>>>(
      hid, DM, W_g, 2048, qkvg + 2 * QKD, QKVG_LD, M, 2048, DM);

  // 2) rotary on q,k ; 3) silu on gate
  rotary_kernel<<<(M * (QKD / 2)) / 256, 256, 0, stream>>>(qkvg);
  silu_kernel<<<(M * VD) / 256, 256, 0, stream>>>(qkvg);

  // 4) retention -> ret (f32, [b,t, h-major V_DIM])
  retention_kernel<<<dim3(T_ / 64, NH * 2, B_), 256, 0, stream>>>(qkvg, decay, ret);

  // 5) curr_kv -> tail of d_out
  currkv_kernel<<<dim3(2, NH, B_), 256, 0, stream>>>(qkvg, decay,
                                                     out + (size_t)M * DM);

  // 6) groupnorm * gate -> ng (bf16)
  gn_gate_kernel<<<M, 256, 0, stream>>>(ret, qkvg, ng);

  // 7) proj_out = ng @ W_proj (f32 out)
  gemm_bf16_kernel<false><<<dim3(DM / 128, M / 128), 256, 0, stream>>>(
      ng, VD, W_proj, DM, out, DM, M, DM, VD);
}
